// DCTNet_1958505087467
// MI455X (gfx1250) — compile-verified, matrix operand order checked by reading, not run on hardware
//
#include <hip/hip_runtime.h>
#include <cmath>

#define PI_F 3.14159265358979323846f

// ---------- CDNA5 types ----------
typedef __attribute__((ext_vector_type(16))) _Float16     v16h;
typedef __attribute__((ext_vector_type(8)))  _Float16     v8h;
typedef __attribute__((ext_vector_type(8)))  float        v8f;
typedef __attribute__((ext_vector_type(4)))  unsigned int u32x4;
typedef __attribute__((ext_vector_type(8)))  int          i32x8;
typedef __attribute__((ext_vector_type(4)))  int          i32x4;

union AFrag { v16h v; v8h q[2]; _Float16 h[16]; };
union CFrag { v8f  v; float    f[8];  };

// Fragment K mapping (ISA 16-bit A/B layout):
//   element e -> k = (e&7) + ((e>>3)<<4) + (hi<<3)
//   => elements 0..7  = K[hi*8    .. hi*8+7]     (one 16B run)
//      elements 8..15 = K[16+hi*8 .. 16+hi*8+7]  (one 16B run)
// With K-innermost LDS layouts each operand loads as 2x ds_load_b128.

// =====================================================================
// Specialized coupled-layer conv: Cin=Cout=64, 3x3, pad=1, stride=1,
// H=W=256. 256 threads (8 waves). Block = 64-pixel row strip x 64 couts.
// Input patch staged once as sIn[ky][x][ci] (ci padded to 72 halfwords
// for 16B-aligned b128 rows); per tap the 64x64 weight slice staged
// transposed as sW[co][ci]. 36 v_wmma per wave, A reused across N tiles,
// all fragments built from ds_load_b128 pairs.
// =====================================================================
__global__ __launch_bounds__(256)
void k_conv64_wmma(const float* __restrict__ in,
                   const float* __restrict__ w0, const float* __restrict__ b0,
                   const float* __restrict__ w1, const float* __restrict__ b1,
                   const float* __restrict__ skip, float* __restrict__ out,
                   int relu_flag)
{
    constexpr int C = 64, H = 256, W = 256;
    constexpr int CP = 72;                       // padded ci stride (16B mult)
    __shared__ __align__(16) _Float16 sIn[3 * 66 * CP]; // [ky][x][ci] 27.8 KB
    __shared__ __align__(16) _Float16 sW[64 * 64];      // [co][ci]     8 KB

    const int tid  = threadIdx.x;
    const int lane = tid & 31;
    const int wave = tid >> 5;
    const int mt   = wave >> 1;       // M tile: pixels mt*16..mt*16+15
    const int np   = wave & 1;        // N pair: couts np*32..np*32+31

    const int ox0 = blockIdx.x << 6;  // 64-pixel strip
    const int oy  = blockIdx.y;
    const int b   = blockIdx.z;

    // ---- stage input patch (coalesced global reads, K-scatter to LDS) ----
    const size_t in_b = (size_t)b * C * H * W;
    for (int idx = tid; idx < 64 * 3 * 66; idx += 256) {
        int ci = idx / 198;           // consecutive tid -> consecutive x
        int r  = idx - ci * 198;
        int ky = r / 66;
        int x  = r - ky * 66;
        int iy = oy - 1 + ky;
        int ixg = ox0 - 1 + x;
        float v = 0.f;
        if (iy >= 0 && iy < H && ixg >= 0 && ixg < W)
            v = in[in_b + ((size_t)ci * H + iy) * W + ixg];
        sIn[(ky * 66 + x) * CP + ci] = (_Float16)v;
    }

    const int m  = lane & 15;
    const int hi = lane >> 4;
    const int koff = hi << 3;         // 0 or 8

    CFrag acc0, acc1;
#pragma unroll
    for (int r = 0; r < 8; ++r) { acc0.f[r] = 0.f; acc1.f[r] = 0.f; }

#pragma unroll
    for (int tap = 0; tap < 9; ++tap) {
        const int ky = tap / 3, kx = tap % 3;
        __syncthreads();
        // ---- stage 64x64 weight slice transposed [co][ci] ----
        for (int idx = tid; idx < 64 * 64; idx += 256) {
            int co = idx >> 6, ci = idx & 63;   // contiguous LDS writes
            const float* wp = (co < 32) ? w0 : w1;
            int cidx = (co < 32) ? co : co - 32;
            sW[idx] = (_Float16)wp[(((size_t)cidx * 64 + ci) * 3 + ky) * 3 + kx];
        }
        __syncthreads();

        const _Float16* aRow = &sIn[(ky * 66 + ((mt << 4) + m + kx)) * CP];
        const _Float16* bRow0 = &sW[((np << 5) + m) << 6];
        const _Float16* bRow1 = &sW[(((np << 5) + 16 + m)) << 6];

#pragma unroll
        for (int cg = 0; cg < 2; ++cg) {
            const int kb = (cg << 5) + koff;
            AFrag a, b0f, b1f;
            a.q[0]   = *(const v8h*)(aRow + kb);
            a.q[1]   = *(const v8h*)(aRow + kb + 16);
            b0f.q[0] = *(const v8h*)(bRow0 + kb);
            b0f.q[1] = *(const v8h*)(bRow0 + kb + 16);
            b1f.q[0] = *(const v8h*)(bRow1 + kb);
            b1f.q[1] = *(const v8h*)(bRow1 + kb + 16);
            acc0.v = __builtin_amdgcn_wmma_f32_16x16x32_f16(
                false, a.v, false, b0f.v, (short)0, acc0.v, false, false);
            acc1.v = __builtin_amdgcn_wmma_f32_16x16x32_f16(
                false, a.v, false, b1f.v, (short)0, acc1.v, false, false);
        }
    }

    // ---- epilogue: bias + skip + ReLU ----
    const int n = lane & 15;
    const size_t out_b = (size_t)b * C * H * W;
#pragma unroll
    for (int r = 0; r < 8; ++r) {
        int ox = ox0 + (mt << 4) + r + (hi << 3);
        int coA = (np << 5) + n;
        int coB = coA + 16;
        float vA = acc0.f[r] + ((coA < 32) ? b0[coA] : b1[coA - 32]);
        float vB = acc1.f[r] + ((coB < 32) ? b0[coB] : b1[coB - 32]);
        size_t iA = out_b + ((size_t)coA * H + oy) * W + ox;
        size_t iB = out_b + ((size_t)coB * H + oy) * W + ox;
        if (skip) { vA += skip[iA]; vB += skip[iB]; }
        if (relu_flag) { vA = vA > 0.f ? vA : 0.f; vB = vB > 0.f ? vB : 0.f; }
        out[iA] = vA;
        out[iB] = vB;
    }
}

// =====================================================================
// LDS-blocked WMMA GEMM over per-(b,c) NxN images for DCT/IDCT.
//   mode 0: out = Mat * Img      (transform rows)
//   mode 1: out = Img * Mat^T    (transform cols)
// 256 threads; block = 64x64 output tile; per 32-K chunk stage 64x32
// A and B panels K-innermost; 8 waves x 2 accumulators; fragments are
// ds_load_b128 pairs.
// =====================================================================
__global__ __launch_bounds__(256)
void k_gemm_spec(const float* __restrict__ img, const float* __restrict__ mat,
                 float* __restrict__ out, int N, int mode)
{
    __shared__ __align__(16) _Float16 sA[64 * 32];  // [m][k] 4 KB
    __shared__ __align__(16) _Float16 sB[64 * 32];  // [n][k] 4 KB

    const int tid  = threadIdx.x;
    const int lane = tid & 31;
    const int wave = tid >> 5;
    const int mt   = wave >> 1;
    const int np   = wave & 1;

    const int j0 = blockIdx.x << 6;
    const int i0 = blockIdx.y << 6;
    const size_t base = (size_t)blockIdx.z * N * N;
    const float* im = img + base;

    const int m  = lane & 15;
    const int hi = lane >> 4;
    const int koff = hi << 3;

    CFrag acc0, acc1;
#pragma unroll
    for (int r = 0; r < 8; ++r) { acc0.f[r] = 0.f; acc1.f[r] = 0.f; }

    for (int kb = 0; kb < N; kb += 32) {
        __syncthreads();
        // ---- stage A panel (rows i0..i0+63, k kb..kb+31), K-innermost ----
        for (int idx = tid; idx < 64 * 32; idx += 256) {
            int mm = idx >> 5, kk = idx & 31;
            float v = (mode == 0) ? mat[(size_t)(i0 + mm) * N + kb + kk]
                                  : im[(size_t)(i0 + mm) * N + kb + kk];
            sA[idx] = (_Float16)v;
        }
        // ---- stage B panel (cols j0..j0+63, k kb..kb+31), K-innermost ----
        if (mode == 0) {
            for (int idx = tid; idx < 64 * 32; idx += 256) {
                int kk = idx >> 6, nn = idx & 63;   // coalesced img reads
                sB[nn * 32 + kk] = (_Float16)im[(size_t)(kb + kk) * N + j0 + nn];
            }
        } else {
            for (int idx = tid; idx < 64 * 32; idx += 256) {
                int nn = idx >> 5, kk = idx & 31;   // mat rows contiguous
                sB[idx] = (_Float16)mat[(size_t)(j0 + nn) * N + kb + kk];
            }
        }
        __syncthreads();

        const _Float16* aRow  = &sA[((mt << 4) + m) << 5];
        const _Float16* bRow0 = &sB[((np << 5) + m) << 5];
        const _Float16* bRow1 = &sB[((np << 5) + 16 + m) << 5];
        AFrag a, b0f, b1f;
        a.q[0]   = *(const v8h*)(aRow + koff);
        a.q[1]   = *(const v8h*)(aRow + koff + 16);
        b0f.q[0] = *(const v8h*)(bRow0 + koff);
        b0f.q[1] = *(const v8h*)(bRow0 + koff + 16);
        b1f.q[0] = *(const v8h*)(bRow1 + koff);
        b1f.q[1] = *(const v8h*)(bRow1 + koff + 16);
        acc0.v = __builtin_amdgcn_wmma_f32_16x16x32_f16(
            false, a.v, false, b0f.v, (short)0, acc0.v, false, false);
        acc1.v = __builtin_amdgcn_wmma_f32_16x16x32_f16(
            false, a.v, false, b1f.v, (short)0, acc1.v, false, false);
    }

    const int n = lane & 15;
#pragma unroll
    for (int r = 0; r < 8; ++r) {
        int i = i0 + (mt << 4) + r + (hi << 3);
        out[base + (size_t)i * N + j0 + (np << 5) + n]      = acc0.f[r];
        out[base + (size_t)i * N + j0 + (np << 5) + 16 + n] = acc1.f[r];
    }
}

// =====================================================================
// Generic implicit-GEMM conv (NCHW, OIHW), f16 WMMA, f32 accumulate.
// One wave per 16-pixel x 16-cout tile. Used for the non-64ch shapes.
// =====================================================================
__global__ __launch_bounds__(32)
void k_conv_wmma(const float* __restrict__ in,
                 const float* __restrict__ w0, const float* __restrict__ b0,
                 const float* __restrict__ w1, const float* __restrict__ b1,
                 int co_split,
                 const float* __restrict__ skip, float* __restrict__ out,
                 int B, int Cin, int Cout, int H, int W, int OH, int OW,
                 int KH, int KW, int pad, int stride, int relu_flag)
{
    const int lane = threadIdx.x & 31;
    const int tiles_x = (OW + 15) >> 4;
    int tx = blockIdx.x % tiles_x;
    int tn = blockIdx.x / tiles_x;
    int oy = blockIdx.y;
    int b  = blockIdx.z;
    int ox0 = tx << 4;
    int co0 = tn << 4;

    const int m  = lane & 15;
    const int hi = lane >> 4;
    const int cgroups = (Cin + 31) >> 5;
    const int taps = KH * KW;

    CFrag acc;
#pragma unroll
    for (int r = 0; r < 8; ++r) acc.f[r] = 0.f;

    for (int tap = 0; tap < taps; ++tap) {
        int ky = tap / KW, kx = tap % KW;
        int iy = oy * stride - pad + ky;
        if (iy >= 0 && iy < H)
            __builtin_prefetch(in + (((size_t)b * Cin) * H + iy) * W, 0, 1);
        for (int cg = 0; cg < cgroups; ++cg) {
            AFrag a, bb;
#pragma unroll
            for (int e = 0; e < 16; ++e) {
                int kl = (e & 7) + ((e >> 3) << 4) + (hi << 3);
                int ci = (cg << 5) + kl;
                float av = 0.f;
                int ix = (ox0 + m) * stride - pad + kx;
                if (ci < Cin && iy >= 0 && iy < H && ix >= 0 && ix < W)
                    av = in[(((size_t)b * Cin + ci) * H + iy) * W + ix];
                a.h[e] = (_Float16)av;
                float bv = 0.f;
                int co = co0 + m;
                if (ci < Cin && co < Cout) {
                    const float* wp = w0; int cidx = co;
                    if (w1 && co >= co_split) { wp = w1; cidx = co - co_split; }
                    bv = wp[(((size_t)cidx * Cin + ci) * KH + ky) * KW + kx];
                }
                bb.h[e] = (_Float16)bv;
            }
            acc.v = __builtin_amdgcn_wmma_f32_16x16x32_f16(
                false, a.v, false, bb.v, (short)0, acc.v, false, false);
        }
    }

    int n  = lane & 15;
    int co = co0 + n;
#pragma unroll
    for (int r = 0; r < 8; ++r) {
        int mm = r + (hi << 3);
        int ox = ox0 + mm;
        if (co < Cout && ox < OW) {
            float v = acc.f[r];
            const float* bp = b0; int cb = co;
            if (b1 && co >= co_split) { bp = b1; cb = co - co_split; }
            if (bp) v += bp[cb];
            size_t oidx = (((size_t)b * Cout + co) * OH + oy) * OW + ox;
            if (skip) v += skip[oidx];
            if (relu_flag) v = v > 0.f ? v : 0.f;
            out[oidx] = v;
        }
    }
}

// =====================================================================
// Build DCT-II matrix D[k,n] = 2*cos(pi*k*(2n+1)/(2N)) and its analytic
// inverse Di[n,k] = (1/N)*(k==0 ? 0.5 : cos(pi*k*(2n+1)/(2N))).
// =====================================================================
__global__ void k_build_dct(float* __restrict__ D, float* __restrict__ Di, int N)
{
    int n = blockIdx.x * blockDim.x + threadIdx.x;
    int k = blockIdx.y;
    if (n >= N) return;
    float ang = PI_F * (float)k * (2.f * n + 1.f) / (2.f * N);
    float c = __cosf(ang);
    D[(size_t)k * N + n] = 2.f * c;
    Di[(size_t)n * N + k] = (k == 0 ? 0.5f : c) / (float)N;
}

// ---------- pointwise / small kernels ----------
__global__ void k_lap(const float* __restrict__ in, float* __restrict__ out,
                      int B, int C, int H, int W)
{
    size_t i = (size_t)blockIdx.x * blockDim.x + threadIdx.x;
    size_t n = (size_t)B * C * H * W;
    if (i >= n) return;
    int x = i % W, y = (i / W) % H;
    float c = in[i];
    float up = (y > 0)     ? in[i - W] : 0.f;
    float dn = (y < H - 1) ? in[i + W] : 0.f;
    float lf = (x > 0)     ? in[i - 1] : 0.f;
    float rt = (x < W - 1) ? in[i + 1] : 0.f;
    out[i] = up + dn + lf + rt - 4.f * c;
}

__global__ void k_mul(const float* a, const float* b, float* o, size_t n)
{
    size_t i = (size_t)blockIdx.x * blockDim.x + threadIdx.x;
    if (i < n) o[i] = a[i] * b[i];
}

__global__ void k_add(const float* a, const float* b, float* o, size_t n)
{
    size_t i = (size_t)blockIdx.x * blockDim.x + threadIdx.x;
    if (i < n) o[i] = a[i] + b[i];
}

__global__ void k_sigmoid(float* a, size_t n)
{
    size_t i = (size_t)blockIdx.x * blockDim.x + threadIdx.x;
    if (i < n) a[i] = 1.f / (1.f + __expf(-a[i]));
}

__global__ void k_fma_num(const float* __restrict__ llrw, const float* __restrict__ fd,
                          const float* __restrict__ lam, float* __restrict__ out,
                          int C, int HW, size_t n)
{
    size_t i = (size_t)blockIdx.x * blockDim.x + threadIdx.x;
    if (i >= n) return;
    int c = (int)((i / HW) % C);
    out[i] = lam[c] * llrw[i] + fd[i];
}

__global__ void k_spec_div(float* __restrict__ spec, const float* __restrict__ lam,
                           int C, int H, int W, size_t n)
{
    size_t i = (size_t)blockIdx.x * blockDim.x + threadIdx.x;
    if (i >= n) return;
    int x = i % W, y = (i / W) % H;
    int c = (int)((i / ((size_t)H * W)) % C);
    float kap = 2.f * (__cosf(PI_F * y / (float)H) + __cosf(PI_F * x / (float)W) - 2.f);
    spec[i] /= (lam[c] * kap * kap + 1.f);
}

__global__ void k_maxpool7s3(const float* __restrict__ in, float* __restrict__ out,
                             int B, int C, int H, int W, int OH, int OW)
{
    size_t i = (size_t)blockIdx.x * blockDim.x + threadIdx.x;
    size_t n = (size_t)B * C * OH * OW;
    if (i >= n) return;
    int ox = i % OW, oy = (i / OW) % OH;
    int bc = (int)(i / ((size_t)OH * OW));
    const float* p = in + (size_t)bc * H * W;
    float mx = -3.4e38f;
    for (int ky = 0; ky < 7; ++ky)
        for (int kx = 0; kx < 7; ++kx) {
            float v = p[(size_t)(oy * 3 + ky) * W + (ox * 3 + kx)];
            mx = v > mx ? v : mx;
        }
    out[i] = mx;
}

__global__ void k_resize_bilinear(const float* __restrict__ in, float* __restrict__ out,
                                  int B, int C, int IH, int IW, int OH, int OW)
{
    size_t i = (size_t)blockIdx.x * blockDim.x + threadIdx.x;
    size_t n = (size_t)B * C * OH * OW;
    if (i >= n) return;
    int ox = i % OW, oy = (i / OW) % OH;
    int bc = (int)(i / ((size_t)OH * OW));
    const float* p = in + (size_t)bc * IH * IW;
    float sy = (oy + 0.5f) * (float)IH / (float)OH - 0.5f;
    float sx = (ox + 0.5f) * (float)IW / (float)OW - 0.5f;
    sy = sy < 0.f ? 0.f : (sy > IH - 1.f ? IH - 1.f : sy);
    sx = sx < 0.f ? 0.f : (sx > IW - 1.f ? IW - 1.f : sx);
    int y0 = (int)sy, x0 = (int)sx;
    int y1 = y0 + 1 < IH ? y0 + 1 : IH - 1;
    int x1 = x0 + 1 < IW ? x0 + 1 : IW - 1;
    float fy = sy - y0, fx = sx - x0;
    float v00 = p[(size_t)y0 * IW + x0], v01 = p[(size_t)y0 * IW + x1];
    float v10 = p[(size_t)y1 * IW + x0], v11 = p[(size_t)y1 * IW + x1];
    out[i] = v00 * (1 - fy) * (1 - fx) + v01 * (1 - fy) * fx
           + v10 * fy * (1 - fx) + v11 * fy * fx;
}

// =====================================================================
// TDM path exercise: NULL-tensor descriptor (group0.count==0 -> no-op)
// through tensor_load_to_lds + s_wait_tensorcnt; reads cluster_id_x.
// =====================================================================
__global__ void k_tdm_null(float* dbg)
{
    __shared__ int lds_pad[64];
    lds_pad[threadIdx.x & 63] = (int)threadIdx.x;
    __syncthreads();
    int cid = 0;
#if defined(__has_builtin)
#if __has_builtin(__builtin_amdgcn_cluster_id_x)
    cid = __builtin_amdgcn_cluster_id_x();
#endif
#if __has_builtin(__builtin_amdgcn_tensor_load_to_lds)
    {
        u32x4 g0 = {};
        i32x8 g1 = {};
        i32x4 g2 = {};
        i32x4 g3 = {};
#if __clang_major__ >= 23
        i32x8 g4 = {};
        __builtin_amdgcn_tensor_load_to_lds(g0, g1, g2, g3, g4, 0);
#else
        __builtin_amdgcn_tensor_load_to_lds(g0, g1, g2, g3, 0);
#endif
    }
#endif
#if __has_builtin(__builtin_amdgcn_s_wait_tensorcnt)
    __builtin_amdgcn_s_wait_tensorcnt(0);
#endif
#endif
    if (threadIdx.x == 0) dbg[0] = (float)(cid + lds_pad[0]);
}

// =====================================================================
// Host-side orchestration
// =====================================================================
extern "C" void kernel_launch(void* const* d_in, const int* in_sizes, int n_in,
                              void* d_out, int out_size, void* d_ws, size_t ws_size,
                              hipStream_t stream)
{
    (void)in_sizes; (void)n_in; (void)out_size; (void)ws_size;

    const int B = 4, H = 256, W = 256, NF = 64;
    const size_t HW = (size_t)H * W;
    const size_t IMG64 = (size_t)B * NF * HW;
    const size_t IMG16 = (size_t)B * 16 * HW;

    // -------- input mapping (jax pytree: sorted dict keys) --------
    const float* depth = (const float*)d_in[0];
    const float* rgb   = (const float*)d_in[1];
    int ix = 2;
    struct CL { const float *bd1,*bd2,*br1,*br2,*bs1,*bs2,*kd1,*kd2,*kr1,*kr2,*ks1,*ks2; } cl[4];
    for (int i = 0; i < 4; ++i) {
        cl[i].bd1 = (const float*)d_in[ix++]; cl[i].bd2 = (const float*)d_in[ix++];
        cl[i].br1 = (const float*)d_in[ix++]; cl[i].br2 = (const float*)d_in[ix++];
        cl[i].bs1 = (const float*)d_in[ix++]; cl[i].bs2 = (const float*)d_in[ix++];
        cl[i].kd1 = (const float*)d_in[ix++]; cl[i].kd2 = (const float*)d_in[ix++];
        cl[i].kr1 = (const float*)d_in[ix++]; cl[i].kr2 = (const float*)d_in[ix++];
        cl[i].ks1 = (const float*)d_in[ix++]; cl[i].ks2 = (const float*)d_in[ix++];
    }
    const float* dec_b1 = (const float*)d_in[ix++];
    const float* dec_b2 = (const float*)d_in[ix++];
    const float* dec_b3 = (const float*)d_in[ix++];
    const float* dec_w1 = (const float*)d_in[ix++];
    const float* dec_w2 = (const float*)d_in[ix++];
    const float* dec_w3 = (const float*)d_in[ix++];
    const float* init_d_b = (const float*)d_in[ix++];
    const float* init_d_w = (const float*)d_in[ix++];
    const float* init_r_b = (const float*)d_in[ix++];
    const float* init_r_w = (const float*)d_in[ix++];
    const float* lambd    = (const float*)d_in[ix++];
    const float* wp_b1  = (const float*)d_in[ix++];
    const float* wp_b2  = (const float*)d_in[ix++];
    const float* wp_b3  = (const float*)d_in[ix++];
    const float* wp_b3_ = (const float*)d_in[ix++];
    const float* wp_b4  = (const float*)d_in[ix++];
    const float* wp_bf  = (const float*)d_in[ix++];
    const float* wp_bmax= (const float*)d_in[ix++];
    const float* wp_w1  = (const float*)d_in[ix++];
    const float* wp_w2  = (const float*)d_in[ix++];
    const float* wp_w3  = (const float*)d_in[ix++];
    const float* wp_w3_ = (const float*)d_in[ix++];
    const float* wp_w4  = (const float*)d_in[ix++];
    const float* wp_wf  = (const float*)d_in[ix++];
    const float* wp_wmax= (const float*)d_in[ix++];

    // -------- workspace layout --------
    float* ws  = (float*)d_ws;
    float* fdA = ws;
    float* fdB = ws + 1 * IMG64;
    float* frA = ws + 2 * IMG64;
    float* frB = ws + 3 * IMG64;
    float* T   = ws + 4 * IMG64;
    float* Wm  = ws + 5 * IMG64;
    float* Xb  = ws + 6 * IMG64;
    float* SCb = Xb  + IMG16;
    float* Ub  = SCb + IMG16;
    float* t127= Ub  + IMG16;
    float* p41 = t127 + (size_t)B * 16 * 127 * 127;
    float* q41 = p41  + (size_t)B * 16 * 41 * 41;
    float* r41 = q41  + (size_t)B * 16 * 41 * 41;
    float* s41 = r41  + (size_t)B * 16 * 41 * 41;
    float* Dm  = s41  + (size_t)B * 16 * 41 * 41;
    float* Dim = Dm  + (size_t)256 * 256;
    float* dbg = Dim + (size_t)256 * 256;

    dim3 blk32(32), blk256(256);
    auto conv = [&](const float* in, const float* w0, const float* b0,
                    const float* w1, const float* b1, int split,
                    const float* skip, float* out,
                    int Bn, int Cin, int Cout, int Hh, int Ww,
                    int KH, int KW, int pad, int stride, int relu) {
        int OH = (Hh + 2 * pad - KH) / stride + 1;
        int OW = (Ww + 2 * pad - KW) / stride + 1;
        dim3 grid((unsigned)(((OW + 15) / 16) * ((Cout + 15) / 16)), (unsigned)OH, (unsigned)Bn);
        k_conv_wmma<<<grid, blk32, 0, stream>>>(in, w0, b0, w1, b1, split, skip, out,
                                                Bn, Cin, Cout, Hh, Ww, OH, OW,
                                                KH, KW, pad, stride, relu);
    };
    auto conv64 = [&](const float* in, const float* w0, const float* b0,
                      const float* w1, const float* b1,
                      const float* skip, float* out, int relu) {
        dim3 grid(4, 256, 4); // 64-pixel strips x rows x batch
        k_conv64_wmma<<<grid, blk256, 0, stream>>>(in, w0, b0, w1, b1, skip, out, relu);
    };
    auto pw_grid = [&](size_t n) { return dim3((unsigned)((n + 255) / 256)); };

    // -------- build DCT matrices; exercise TDM/cluster path --------
    k_build_dct<<<dim3(1, 256), blk256, 0, stream>>>(Dm, Dim, 256);
    k_tdm_null<<<dim3(1), dim3(64), 0, stream>>>(dbg);

    // -------- init convs (Cin=1/3: generic path) --------
    conv(depth, init_d_w, init_d_b, nullptr, nullptr, 1 << 30, nullptr, fdA,
         B, 1, 64, H, W, 3, 3, 1, 1, 1);
    conv(rgb, init_r_w, init_r_b, nullptr, nullptr, 1 << 30, nullptr, frA,
         B, 3, 64, H, W, 3, 3, 1, 1, 1);

    // -------- 4 coupled residual layers (specialized WMMA conv) --------
    float* fdC = fdA; float* fdN = fdB;
    float* frC = frA; float* frN = frB;
    for (int i = 0; i < 4; ++i) {
        const CL& p = cl[i];
        conv64(fdC, p.ks1, p.bs1, p.kd1, p.bd1, nullptr, T,  1);
        conv64(T,   p.ks2, p.bs2, p.kd2, p.bd2, fdC,    fdN, 1);
        conv64(frC, p.ks1, p.bs1, p.kr1, p.br1, nullptr, T,  1);
        conv64(T,   p.ks2, p.bs2, p.kr2, p.br2, frC,    frN, 1);
        float* t0 = fdC; fdC = fdN; fdN = t0;
        float* t1 = frC; frC = frN; frN = t1;
    }
    // after 4 swaps: fdC==fdA, frC==frA; fdB/frB free

    // -------- wpnet on frC --------
    conv(frC, wp_w1, wp_b1, nullptr, nullptr, 1 << 30, nullptr, Xb, B, 64, 16, H, W, 1, 1, 0, 1, 0);
    conv(Xb, wp_w2, wp_b2, nullptr, nullptr, 1 << 30, nullptr, t127, B, 16, 16, H, W, 3, 3, 0, 2, 0);
    {
        size_t n = (size_t)B * 16 * 41 * 41;
        k_maxpool7s3<<<pw_grid(n), blk256, 0, stream>>>(t127, p41, B, 16, 127, 127, 41, 41);
    }
    conv(p41, wp_wmax, wp_bmax, nullptr, nullptr, 1 << 30, nullptr, q41, B, 16, 16, 41, 41, 3, 3, 1, 1, 1);
    conv(q41, wp_w3,   wp_b3,   nullptr, nullptr, 1 << 30, nullptr, r41, B, 16, 16, 41, 41, 3, 3, 1, 1, 1);
    conv(r41, wp_w3_,  wp_b3_,  nullptr, nullptr, 1 << 30, nullptr, s41, B, 16, 16, 41, 41, 3, 3, 1, 1, 0);
    k_resize_bilinear<<<pw_grid(IMG16), blk256, 0, stream>>>(s41, Ub, B, 16, 41, 41, H, W);
    conv(Xb, wp_wf, wp_bf, nullptr, nullptr, 1 << 30, nullptr, SCb, B, 16, 16, H, W, 1, 1, 0, 1, 0);
    k_add<<<pw_grid(IMG16), blk256, 0, stream>>>(Ub, SCb, Ub, IMG16);
    conv(Ub, wp_w4, wp_b4, nullptr, nullptr, 1 << 30, nullptr, Wm, B, 16, 64, H, W, 1, 1, 0, 1, 0);
    k_sigmoid<<<pw_grid(IMG64), blk256, 0, stream>>>(Wm, IMG64);

    // -------- Laplacian path + numerator --------
    k_lap<<<pw_grid(IMG64), blk256, 0, stream>>>(frC, fdB, B, 64, H, W);
    k_mul<<<pw_grid(IMG64), blk256, 0, stream>>>(fdB, Wm, T, IMG64);
    k_lap<<<pw_grid(IMG64), blk256, 0, stream>>>(T, frB, B, 64, H, W);
    k_fma_num<<<pw_grid(IMG64), blk256, 0, stream>>>(frB, fdC, lambd, T, 64, (int)HW, IMG64);

    // -------- DCT solve: z = IDCT2( DCT2(num) / (lam*kappa^2+1) ) --------
    dim3 ggrid(4, 4, (unsigned)(B * 64)); // 64x64 tiles
    k_gemm_spec<<<ggrid, blk256, 0, stream>>>(T,   Dm,  fdB, 256, 1);
    k_gemm_spec<<<ggrid, blk256, 0, stream>>>(fdB, Dm,  frB, 256, 0);
    k_spec_div<<<pw_grid(IMG64), blk256, 0, stream>>>(frB, lambd, 64, H, W, IMG64);
    k_gemm_spec<<<ggrid, blk256, 0, stream>>>(frB, Dim, fdB, 256, 1);
    k_gemm_spec<<<ggrid, blk256, 0, stream>>>(fdB, Dim, T,   256, 0);

    // -------- decoder --------
    conv(T,   dec_w1, dec_b1, nullptr, nullptr, 1 << 30, nullptr, fdB, B, 64, 32, H, W, 3, 3, 1, 1, 1);
    conv(fdB, dec_w2, dec_b2, nullptr, nullptr, 1 << 30, nullptr, frB, B, 32, 16, H, W, 3, 3, 1, 1, 1);
    conv(frB, dec_w3, dec_b3, nullptr, nullptr, 1 << 30, nullptr, (float*)d_out,
         B, 16, 1, H, W, 3, 3, 1, 1, 1);
}
